// CSGO_model_3083786518630
// MI455X (gfx1250) — matrix operand order repaired and checked by reading, not compiled, not run on hardware
//
#include <hip/hip_runtime.h>
#include <math.h>

typedef __attribute__((ext_vector_type(8)))  __bf16 v8bf;
typedef __attribute__((ext_vector_type(16))) __bf16 v16bf;
typedef __attribute__((ext_vector_type(8)))  float  v8f;

#define BK 32      // K step (matches 16x16x32 bf16 WMMA)

__device__ __forceinline__ float gelu_exact(float x) {
    // matches jax.nn.gelu(approximate=False)
    return 0.5f * x * (1.0f + erff(x * 0.70710678118654752f));
}

// C[z] = act(A[z] @ B[z] + bias[z]) (+ res[z])
// A: [M,K] f32 row-major, B: [K,N] f32 row-major, C: [M,N] f32.
// M % BM == 0, N % BN == 0, K % 32 == 0 (guaranteed by launcher).
// ACT: 0=none, 1=gelu(exact), 2=relu. Residual stride = sC.
template<int BM, int BN, int WM, int WN, int ACT, bool HAS_BIAS, bool HAS_RES>
__global__ __launch_bounds__((BM / WM) * (BN / WN) * 32)
void gemm_bf16_wmma(const float* __restrict__ A, const float* __restrict__ Bm,
                    const float* __restrict__ bias, const float* __restrict__ res,
                    float* __restrict__ C,
                    int M, int N, int K,
                    long long sA, long long sB, long long sBias, long long sC)
{
    constexpr int THREADS = (BM / WM) * (BN / WN) * 32;
    constexpr int MI = WM / 16;          // M fragments per wave
    constexpr int NI = WN / 16;          // N fragments per wave
    constexpr int WPR = BN / WN;         // waves per block row
    constexpr int LDA_ = BK + 8;         // 80 B row stride, 16B aligned
    constexpr int LDB_ = BN + 8;         // 16B-aligned row stride
    constexpr int A4 = BM * 8 / THREADS; // float4 loads per thread (A tile)
    constexpr int B4 = BN * 8 / THREADS; // float4 loads per thread (B tile)
    constexpr int BROW4 = BN / 4;        // float4 per B row

    __shared__ __bf16 Al[BM][LDA_];
    __shared__ __bf16 Bl[BK][LDB_];

    const int z = blockIdx.z;
    A += (size_t)z * sA;
    Bm += (size_t)z * sB;
    if (HAS_BIAS) bias += (size_t)z * sBias;
    if (HAS_RES)  res  += (size_t)z * sC;
    C += (size_t)z * sC;

    const int bn = blockIdx.x * BN;
    const int bm = blockIdx.y * BM;

    const int tid    = threadIdx.x;
    const int wave   = tid >> 5;
    const int lane   = tid & 31;
    const int lane16 = lane & 15;
    const int laneHi = lane >> 4;
    const int wr = (wave / WPR) * WM;    // wave M offset within block
    const int wc = (wave % WPR) * WN;    // wave N offset within block

    const v8f vzero = {0.f,0.f,0.f,0.f,0.f,0.f,0.f,0.f};
    v8f acc[MI][NI];
    #pragma unroll
    for (int mi = 0; mi < MI; mi++)
        #pragma unroll
        for (int ni = 0; ni < NI; ni++) acc[mi][ni] = vzero;

    // ---- register-staged global prefetch of tile ks=0
    float4 ra[A4], rb[B4];
    #pragma unroll
    for (int i = 0; i < A4; i++) {
        int f = tid + i * THREADS;
        int r = f >> 3, c4 = f & 7;
        ra[i] = ((const float4*)(A + (size_t)(bm + r) * K))[c4];
    }
    #pragma unroll
    for (int i = 0; i < B4; i++) {
        int f = tid + i * THREADS;
        int r = f / BROW4, c4 = f % BROW4;
        rb[i] = ((const float4*)(Bm + (size_t)r * N + bn))[c4];
    }

    for (int ks = 0; ks < K; ks += BK) {
        // ---- store staged tile (f32 -> bf16) into LDS
        #pragma unroll
        for (int i = 0; i < A4; i++) {
            int f = tid + i * THREADS;
            int r = f >> 3, c4 = f & 7;
            __bf16* dst = &Al[r][c4 * 4];
            dst[0] = (__bf16)ra[i].x; dst[1] = (__bf16)ra[i].y;
            dst[2] = (__bf16)ra[i].z; dst[3] = (__bf16)ra[i].w;
        }
        #pragma unroll
        for (int i = 0; i < B4; i++) {
            int f = tid + i * THREADS;
            int r = f / BROW4, c4 = f % BROW4;
            __bf16* dst = &Bl[r][c4 * 4];
            dst[0] = (__bf16)rb[i].x; dst[1] = (__bf16)rb[i].y;
            dst[2] = (__bf16)rb[i].z; dst[3] = (__bf16)rb[i].w;
        }
        __syncthreads();

        // ---- issue next tile's global loads; latency hides behind WMMAs
        if (ks + BK < K) {
            const int ks2 = ks + BK;
            #pragma unroll
            for (int i = 0; i < A4; i++) {
                int f = tid + i * THREADS;
                int r = f >> 3, c4 = f & 7;
                ra[i] = ((const float4*)(A + (size_t)(bm + r) * K + ks2))[c4];
            }
            #pragma unroll
            for (int i = 0; i < B4; i++) {
                int f = tid + i * THREADS;
                int r = f / BROW4, c4 = f % BROW4;
                rb[i] = ((const float4*)(Bm + (size_t)(ks2 + r) * N + bn))[c4];
            }
        }

        // ---- load fragments (ISA 7.12.2 layouts) and issue MI*NI WMMAs
        v16bf af[MI], bfr[NI];
        #pragma unroll
        for (int mi = 0; mi < MI; mi++) {
            // A 16x32: lane m = lane16; elems 0..7 -> K = laneHi*8 + 0..7,
            //          elems 8..15 -> K = 16 + laneHi*8 + 0..7
            const v8bf* p = (const v8bf*)&Al[wr + mi * 16 + lane16][laneHi * 8];
            af[mi] = __builtin_shufflevector(p[0], p[2],
                0,1,2,3,4,5,6,7,8,9,10,11,12,13,14,15);
        }
        #pragma unroll
        for (int ni = 0; ni < NI; ni++) {
            // B 32x16: lane n = lane16; elems 0..7 -> K = laneHi*8 + 0..7,
            //          elems 8..15 -> K = 16 + laneHi*8 + 0..7
            #pragma unroll
            for (int s = 0; s < 16; s++)
                bfr[ni][s] = Bl[(s < 8 ? 0 : 8) + laneHi * 8 + s][wc + ni * 16 + lane16];
        }
        #pragma unroll
        for (int mi = 0; mi < MI; mi++)
            #pragma unroll
            for (int ni = 0; ni < NI; ni++)
                acc[mi][ni] = __builtin_amdgcn_wmma_f32_16x16x32_bf16(
                    false, af[mi], false, bfr[ni],
                    (short)0, acc[mi][ni], false, false);
        __syncthreads();
    }

    // ---- straight-line epilogue: bias + activation (+ residual), f32 out
    #pragma unroll
    for (int ni = 0; ni < NI; ni++) {
        const int col = bn + wc + ni * 16 + lane16;
        const float bv = HAS_BIAS ? bias[col] : 0.f;
        #pragma unroll
        for (int mi = 0; mi < MI; mi++) {
            #pragma unroll
            for (int v = 0; v < 8; v++) {
                int row = bm + wr + mi * 16 + laneHi * 8 + v;
                float val = acc[mi][ni][v] + bv;
                if constexpr (ACT == 1) val = gelu_exact(val);
                if constexpr (ACT == 2) val = fmaxf(val, 0.f);
                size_t idx = (size_t)row * N + col;
                if constexpr (HAS_RES) val += res[idx];
                C[idx] = val;
            }
        }
    }
}

// LayerNorm over rows of width 1024. dbl=1: ln(eps=1e-6, no affine) then
// ln(eps=1e-5, affine). dbl=0: single affine ln (eps=1e-5).
__global__ __launch_bounds__(256)
void ln_kernel(const float* __restrict__ x, const float* __restrict__ g,
               const float* __restrict__ b, float* __restrict__ y, int dbl)
{
    __shared__ float red[256];
    const int row = blockIdx.x;
    const int t = threadIdx.x;
    const float* xr = x + (size_t)row * 1024;
    float v[4];
    #pragma unroll
    for (int i = 0; i < 4; i++) v[i] = xr[t + 256 * i];

    const int npass = dbl ? 2 : 1;
    for (int p = 0; p < npass; p++) {
        float s = 0.f, sq = 0.f;
        #pragma unroll
        for (int i = 0; i < 4; i++) { s += v[i]; sq += v[i] * v[i]; }
        red[t] = s; __syncthreads();
        for (int o = 128; o > 0; o >>= 1) { if (t < o) red[t] += red[t + o]; __syncthreads(); }
        float mean = red[0] * (1.f / 1024.f); __syncthreads();
        red[t] = sq; __syncthreads();
        for (int o = 128; o > 0; o >>= 1) { if (t < o) red[t] += red[t + o]; __syncthreads(); }
        float var = red[0] * (1.f / 1024.f) - mean * mean; __syncthreads();
        float eps = (dbl && p == 0) ? 1e-6f : 1e-5f;
        float rs = rsqrtf(var + eps);
        #pragma unroll
        for (int i = 0; i < 4; i++) v[i] = (v[i] - mean) * rs;
    }
    float* yr = y + (size_t)row * 1024;
    #pragma unroll
    for (int i = 0; i < 4; i++) { int c = t + 256 * i; yr[c] = v[i] * g[c] + b[c]; }
}

// Per-(batch,head) attention: T=32, HD=64. One wave per (b,h); qkv layout
// [token][3*1024] with Q at col h*64, K at 1024+h*64, V at 2048+h*64.
__global__ __launch_bounds__(32)
void attn_kernel(const float* __restrict__ qkv, float* __restrict__ o)
{
    __shared__ float Qs[32][68], Ks[32][68], Vs[32][68];
    __shared__ float Ss[32][33];
    const int bh = blockIdx.x;
    const int bi = bh >> 4, h = bh & 15;
    const int t  = threadIdx.x;           // query row (= lane)
    const int token = bi * 32 + t;
    const float* base = qkv + (size_t)token * 3072 + h * 64;
    #pragma unroll
    for (int d4 = 0; d4 < 16; d4++) {
        float4 q  = ((const float4*)(base       ))[d4];
        float4 k  = ((const float4*)(base + 1024))[d4];
        float4 vv = ((const float4*)(base + 2048))[d4];
        *((float4*)&Qs[t][d4 * 4]) = q;
        *((float4*)&Ks[t][d4 * 4]) = k;
        *((float4*)&Vs[t][d4 * 4]) = vv;
    }
    __syncthreads();
    float mx = -3.4e38f;
    for (int u = 0; u < 32; u++) {
        float s = 0.f;
        for (int d = 0; d < 64; d++) s += Qs[t][d] * Ks[u][d];
        s *= 0.125f;                       // HD^-0.5
        Ss[t][u] = s;
        mx = fmaxf(mx, s);
    }
    float sum = 0.f;
    for (int u = 0; u < 32; u++) { float e = expf(Ss[t][u] - mx); Ss[t][u] = e; sum += e; }
    float inv = 1.f / sum;
    float* orow = o + (size_t)token * 1024 + h * 64;
    for (int d = 0; d < 64; d++) {
        float a = 0.f;
        for (int u = 0; u < 32; u++) a += Ss[t][u] * Vs[u][d];
        orow[d] = a * inv;
    }
}

// x_comb[b] = concat(x[b,0,:], x[b,1,:])  -> [64, 2048]
__global__ __launch_bounds__(256)
void xcomb_kernel(const float* __restrict__ x, float* __restrict__ xc)
{
    int i = blockIdx.x * 256 + threadIdx.x;   // 64*2048 total
    int b = i >> 11, j = i & 2047;
    float v = (j < 1024) ? x[(size_t)(b * 32) * 1024 + j]
                         : x[(size_t)(b * 32 + 1) * 1024 + (j - 1024)];
    xc[i] = v;
}

// out[b,n] = h2[n][b] . w3[n] + b3[n]   (h2: [26][64][512])
__global__ __launch_bounds__(128)
void head_out_kernel(const float* __restrict__ h2, const float* __restrict__ w3,
                     const float* __restrict__ b3, float* __restrict__ out)
{
    int gid  = blockIdx.x * 4 + (threadIdx.x >> 5);   // 0..1663 = n*64+b
    int lane = threadIdx.x & 31;
    int b = gid & 63, n = gid >> 6;
    const float* hr = h2 + ((size_t)n * 64 + b) * 512;
    const float* w  = w3 + (size_t)n * 512;
    float s = 0.f;
    for (int i = lane; i < 512; i += 32) s += hr[i] * w[i];
    #pragma unroll
    for (int off = 16; off; off >>= 1) s += __shfl_xor(s, off, 32);
    if (lane == 0) out[b * 26 + n] = s + b3[n];
}

extern "C" void kernel_launch(void* const* d_in, const int* in_sizes, int n_in,
                              void* d_out, int out_size, void* d_ws, size_t ws_size,
                              hipStream_t stream)
{
    (void)in_sizes; (void)n_in; (void)out_size; (void)ws_size;
    const float* x      = (const float*)d_in[0];
    const float* qkv_w  = (const float*)d_in[1];
    const float* out_w  = (const float*)d_in[2];
    const float* out_b  = (const float*)d_in[3];
    const float* attn_g = (const float*)d_in[4];
    const float* attn_b = (const float*)d_in[5];
    const float* ff_g   = (const float*)d_in[6];
    const float* ff_b   = (const float*)d_in[7];
    const float* ff_w1  = (const float*)d_in[8];
    const float* ff_b1  = (const float*)d_in[9];
    const float* ff_w2  = (const float*)d_in[10];
    const float* ff_b2  = (const float*)d_in[11];
    const float* hw1    = (const float*)d_in[12];
    const float* hb1    = (const float*)d_in[13];
    const float* hw2    = (const float*)d_in[14];
    const float* hb2    = (const float*)d_in[15];
    const float* hw3    = (const float*)d_in[16];
    const float* hb3    = (const float*)d_in[17];
    float* out = (float*)d_out;

    float* ws     = (float*)d_ws;
    float* xbuf   = ws;                             // 2048*1024 residual
    float* ybuf   = xbuf   + (size_t)2048 * 1024;   // normalized activations
    float* qkvbuf = ybuf   + (size_t)2048 * 1024;   // 2048*3072
    float* obuf   = qkvbuf + (size_t)2048 * 3072;   // attn output 2048*1024
    float* ffbuf  = obuf   + (size_t)2048 * 1024;   // 2048*2048
    float* xcomb  = ffbuf  + (size_t)2048 * 2048;   // 64*2048
    float* h1buf  = xcomb  + (size_t)64 * 2048;     // 26*64*512
    float* h2buf  = h1buf  + (size_t)26 * 64 * 512; // 26*64*512

    hipMemcpyAsync(xbuf, x, (size_t)2048 * 1024 * sizeof(float),
                   hipMemcpyDeviceToDevice, stream);

    for (int l = 0; l < 4; l++) {
        // y = ln_affine(ln_noaffine(x))
        ln_kernel<<<2048, 256, 0, stream>>>(xbuf, attn_g + l * 1024,
                                            attn_b + l * 1024, ybuf, 1);
        // qkv = y @ qkv_w[l]   (2048 x 3072)
        gemm_bf16_wmma<128,128,64,64,0,false,false>
            <<<dim3(3072 / 128, 2048 / 128, 1), 256, 0, stream>>>(
            ybuf, qkv_w + (size_t)l * 1024 * 3072, nullptr, nullptr, qkvbuf,
            2048, 3072, 1024, 0, 0, 0, 0);
        // attention per (b,h)
        attn_kernel<<<1024, 32, 0, stream>>>(qkvbuf, obuf);
        // x = x + (o @ out_w[l] + out_b[l])
        gemm_bf16_wmma<128,128,64,64,0,true,true>
            <<<dim3(1024 / 128, 2048 / 128, 1), 256, 0, stream>>>(
            obuf, out_w + (size_t)l * 1024 * 1024, out_b + l * 1024, xbuf, xbuf,
            2048, 1024, 1024, 0, 0, 0, 0);
        // y = ln_affine(x)
        ln_kernel<<<2048, 256, 0, stream>>>(xbuf, ff_g + l * 1024,
                                            ff_b + l * 1024, ybuf, 0);
        // u = gelu(y @ ff_w1[l] + ff_b1[l])
        gemm_bf16_wmma<128,128,64,64,1,true,false>
            <<<dim3(2048 / 128, 2048 / 128, 1), 256, 0, stream>>>(
            ybuf, ff_w1 + (size_t)l * 1024 * 2048, ff_b1 + l * 2048, nullptr, ffbuf,
            2048, 2048, 1024, 0, 0, 0, 0);
        // x = x + (u @ ff_w2[l] + ff_b2[l])
        gemm_bf16_wmma<128,128,64,64,0,true,true>
            <<<dim3(1024 / 128, 2048 / 128, 1), 256, 0, stream>>>(
            ffbuf, ff_w2 + (size_t)l * 2048 * 1024, ff_b2 + l * 1024, xbuf, xbuf,
            2048, 1024, 2048, 0, 0, 0, 0);
    }

    // heads
    xcomb_kernel<<<512, 256, 0, stream>>>(xbuf, xcomb);
    // h1[n] = relu(xcomb @ hw1[n] + hb1[n])   (A shared: strideA = 0)
    gemm_bf16_wmma<64,64,32,32,2,true,false>
        <<<dim3(512 / 64, 1, 26), 128, 0, stream>>>(
        xcomb, hw1, hb1, nullptr, h1buf,
        64, 512, 2048, 0, (long long)2048 * 512, 512, (long long)64 * 512);
    // h2[n] = relu(h1[n] @ hw2[n] + hb2[n])
    gemm_bf16_wmma<64,64,32,32,2,true,false>
        <<<dim3(512 / 64, 1, 26), 128, 0, stream>>>(
        h1buf, hw2, hb2, nullptr, h2buf,
        64, 512, 512, (long long)64 * 512, (long long)512 * 512, 512,
        (long long)64 * 512);
    // out[b,n] = h2[n][b] . hw3[n] + hb3[n]
    head_out_kernel<<<416, 128, 0, stream>>>(h2buf, hw3, hb3, out);
}
